// RelationLoss_64089501991079
// MI455X (gfx1250) — matrix operand order repaired and checked by reading, not compiled, not run on hardware
//
#include <hip/hip_runtime.h>
#include <math.h>

#define NB   512
#define GSZ  128
#define DIM  256
#define TEMP_INV 10.0f
#define WEIGHT   0.1f
#define EPSN     1e-12f
#define LDS_STRIDE 40            // bf16 elements per staged row (80B -> 20-word bank stride)
#define PBUF (GSZ * LDS_STRIDE)  // one staged 128x32 bf16 tile (10240 B)

typedef __attribute__((ext_vector_type(16))) __bf16 v16bf;
typedef __attribute__((ext_vector_type(8)))  __bf16 v8bf;
typedef __attribute__((ext_vector_type(8)))  float  v8f;

// ---------------- MSE (writes out[0], clears poison) ----------------
__global__ void mse_kernel(const float* __restrict__ preds,
                           const float* __restrict__ y,
                           float* __restrict__ out, int n) {
    __shared__ float red[256];
    float s = 0.f;
    for (int i = threadIdx.x; i < n; i += 256) {
        float d = preds[i] - y[i];
        s += d * d;
    }
    red[threadIdx.x] = s;
    __syncthreads();
    for (int o = 128; o > 0; o >>= 1) {
        if ((int)threadIdx.x < o) red[threadIdx.x] += red[threadIdx.x + o];
        __syncthreads();
    }
    if (threadIdx.x == 0) out[0] = red[0];
}

// ---------------- per-row inverse L2 norms (one wave per row) ----------------
__global__ void rownorm_kernel(const float* __restrict__ rfeat,
                               const float* __restrict__ pfeat,
                               float* __restrict__ invr,
                               float* __restrict__ invp) {
    const int NROWS = NB * GSZ;
    int wid  = blockIdx.x * (blockDim.x >> 5) + (threadIdx.x >> 5);
    int lane = threadIdx.x & 31;
    if (wid >= 2 * NROWS) return;
    const float* src = (wid < NROWS) ? rfeat : pfeat;
    int row = (wid < NROWS) ? wid : wid - NROWS;
    const float* x = src + (size_t)row * DIM;
    float s = 0.f;
    #pragma unroll
    for (int i = 0; i < DIM / 32; ++i) {
        float v = x[lane + i * 32];
        s += v * v;
    }
    #pragma unroll
    for (int off = 16; off > 0; off >>= 1) s += __shfl_xor(s, off, 32);
    if (lane == 0) {
        float n = sqrtf(s);
        float* dst = (wid < NROWS) ? invr : invp;
        dst[row] = 1.0f / fmaxf(n, EPSN);
    }
}

// ---------------- main: per-batch 128x128 Gram via bf16 WMMA + row LSE ----------------
__global__ __launch_bounds__(256) void contrastive_kernel(
        const float* __restrict__ rfeat, const float* __restrict__ pfeat,
        const float* __restrict__ invr,  const float* __restrict__ invp,
        float* __restrict__ ce_part) {
    __shared__ __bf16 ldsP[2 * PBUF];   // double-buffered 128x32 bf16 tiles

    const int b    = blockIdx.x;
    const int tid  = threadIdx.x;
    const int w    = tid >> 5;       // wave 0..7 -> sim rows 16w..16w+15
    const int lane = tid & 31;
    const int half = lane >> 4;
    const int ln   = lane & 15;

    const float* Rb = rfeat + (size_t)b * GSZ * DIM;
    const float* Pb = pfeat + (size_t)b * GSZ * DIM;

    // A fragment source (ISA 16-bit A 16x32 layout):
    // lane l<16 -> row ln, K = {0..7, 16..23}; lane>=16 -> K = {8..15, 24..31}
    const float* aRow = Rb + (size_t)(w * 16 + ln) * DIM;
    const int koff = half * 8;

    // cooperative p-stage assignment: thread -> (row, 16-float chunk)
    const int prow   = tid >> 1;
    const int pchunk = (tid & 1) * 16;
    const float* pSrc = Pb + (size_t)prow * DIM + pchunk;
    __bf16* pDstBase = ldsP + prow * LDS_STRIDE + pchunk;

    v8f acc[8] = {};

    // pipeline registers
    float4 f0, f1, f2, f3;          // staged p slice (raw f32)
    float4 a0, a1, a2, a3;          // staged A slice (raw f32)
    v16bf a_cur;

    // ---- prologue: tile 0 ----
    {
        const float4* ps = (const float4*)(pSrc);
        f0 = ps[0]; f1 = ps[1]; f2 = ps[2]; f3 = ps[3];
        const float4* pa0 = (const float4*)(aRow + koff);
        const float4* pa1 = (const float4*)(aRow + koff + 16);
        a0 = pa0[0]; a1 = pa0[1]; a2 = pa1[0]; a3 = pa1[1];
        // convert & store p tile 0 into buffer 0
        v8bf c0, c1;
        c0[0]=(__bf16)f0.x; c0[1]=(__bf16)f0.y; c0[2]=(__bf16)f0.z; c0[3]=(__bf16)f0.w;
        c0[4]=(__bf16)f1.x; c0[5]=(__bf16)f1.y; c0[6]=(__bf16)f1.z; c0[7]=(__bf16)f1.w;
        c1[0]=(__bf16)f2.x; c1[1]=(__bf16)f2.y; c1[2]=(__bf16)f2.z; c1[3]=(__bf16)f2.w;
        c1[4]=(__bf16)f3.x; c1[5]=(__bf16)f3.y; c1[6]=(__bf16)f3.z; c1[7]=(__bf16)f3.w;
        v8bf* dst = (v8bf*)(pDstBase);
        dst[0] = c0; dst[1] = c1;
        // convert A tile 0
        a_cur[0] =(__bf16)a0.x; a_cur[1] =(__bf16)a0.y; a_cur[2] =(__bf16)a0.z; a_cur[3] =(__bf16)a0.w;
        a_cur[4] =(__bf16)a1.x; a_cur[5] =(__bf16)a1.y; a_cur[6] =(__bf16)a1.z; a_cur[7] =(__bf16)a1.w;
        a_cur[8] =(__bf16)a2.x; a_cur[9] =(__bf16)a2.y; a_cur[10]=(__bf16)a2.z; a_cur[11]=(__bf16)a2.w;
        a_cur[12]=(__bf16)a3.x; a_cur[13]=(__bf16)a3.y; a_cur[14]=(__bf16)a3.z; a_cur[15]=(__bf16)a3.w;
    }
    __syncthreads();

    #pragma unroll
    for (int kt = 0; kt < 8; ++kt) {
        const int cur = kt & 1;
        const int k_next = (kt + 1) * 32;

        // (1) issue next-tile global loads up front (latency hides behind WMMA chain)
        if (kt < 7) {
            const float4* ps = (const float4*)(pSrc + k_next);
            f0 = ps[0]; f1 = ps[1]; f2 = ps[2]; f3 = ps[3];
            const float4* pa0 = (const float4*)(aRow + k_next + koff);
            const float4* pa1 = (const float4*)(aRow + k_next + koff + 16);
            a0 = pa0[0]; a1 = pa0[1]; a2 = pa1[0]; a3 = pa1[1];
        }

        // (2) WMMA chain over 8 column tiles, B fragment prefetched one ahead
        const __bf16* bBase = ldsP + cur * PBUF + ln * LDS_STRIDE + half * 8;
        v8bf pb0 = ((const v8bf*)bBase)[0];
        v8bf pb1 = ((const v8bf*)bBase)[2];
        #pragma unroll
        for (int nt = 0; nt < 8; ++nt) {
            v8bf nb0 = pb0, nb1 = pb1;
            if (nt < 7) {
                const v8bf* nb = (const v8bf*)(bBase + (nt + 1) * 16 * LDS_STRIDE);
                pb0 = nb[0]; pb1 = nb[2];
            }
            v16bf bv = __builtin_shufflevector(nb0, nb1,
                        0,1,2,3,4,5,6,7,8,9,10,11,12,13,14,15);
            acc[nt] = __builtin_amdgcn_wmma_f32_16x16x32_bf16(
                false, a_cur, false, bv, (short)0, acc[nt], false, false);
        }

        // (3) convert + store next tile into the other buffer
        if (kt < 7) {
            v8bf c0, c1;
            c0[0]=(__bf16)f0.x; c0[1]=(__bf16)f0.y; c0[2]=(__bf16)f0.z; c0[3]=(__bf16)f0.w;
            c0[4]=(__bf16)f1.x; c0[5]=(__bf16)f1.y; c0[6]=(__bf16)f1.z; c0[7]=(__bf16)f1.w;
            c1[0]=(__bf16)f2.x; c1[1]=(__bf16)f2.y; c1[2]=(__bf16)f2.z; c1[3]=(__bf16)f2.w;
            c1[4]=(__bf16)f3.x; c1[5]=(__bf16)f3.y; c1[6]=(__bf16)f3.z; c1[7]=(__bf16)f3.w;
            v8bf* dst = (v8bf*)(pDstBase + (1 - cur) * PBUF);
            dst[0] = c0; dst[1] = c1;
            a_cur[0] =(__bf16)a0.x; a_cur[1] =(__bf16)a0.y; a_cur[2] =(__bf16)a0.z; a_cur[3] =(__bf16)a0.w;
            a_cur[4] =(__bf16)a1.x; a_cur[5] =(__bf16)a1.y; a_cur[6] =(__bf16)a1.z; a_cur[7] =(__bf16)a1.w;
            a_cur[8] =(__bf16)a2.x; a_cur[9] =(__bf16)a2.y; a_cur[10]=(__bf16)a2.z; a_cur[11]=(__bf16)a2.w;
            a_cur[12]=(__bf16)a3.x; a_cur[13]=(__bf16)a3.y; a_cur[14]=(__bf16)a3.z; a_cur[15]=(__bf16)a3.w;
        }
        __syncthreads();  // readers of buf[cur] done before it is overwritten next iter
    }

    // ---- epilogue: scale by inv norms / temperature, row-wise logsumexp - diag ----
    const float* invRb = invr + b * GSZ;
    const float* invPb = invp + b * GSZ;
    float ipn[8], irn[8];
    #pragma unroll
    for (int nt = 0; nt < 8; ++nt) ipn[nt] = invPb[nt * 16 + ln];
    #pragma unroll
    for (int e = 0; e < 8; ++e)  irn[e] = invRb[w * 16 + e + 8 * half];

    float ce = 0.0f;
    #pragma unroll
    for (int e = 0; e < 8; ++e) {
        // C layout: acc[nt][e] @ lane l = S[16w + e + 8*(l>=16)][16nt + l%16]
        float v[8];
        float m = -3.402823466e38f;
        #pragma unroll
        for (int nt = 0; nt < 8; ++nt) {
            v[nt] = acc[nt][e] * irn[e] * ipn[nt] * TEMP_INV;
            m = fmaxf(m, v[nt]);
        }
        #pragma unroll
        for (int off = 1; off < 16; off <<= 1) m = fmaxf(m, __shfl_xor(m, off, 32));
        float s = 0.f;
        #pragma unroll
        for (int nt = 0; nt < 8; ++nt) s += __expf(v[nt] - m);
        #pragma unroll
        for (int off = 1; off < 16; off <<= 1) s += __shfl_xor(s, off, 32);
        float lse = m + __logf(s);
        // diag for this row (local row rr = e + 8*half, column tile nt = w, col-in-tile = rr)
        if (ln == e + 8 * half) ce += lse - v[w];
    }
    #pragma unroll
    for (int off = 16; off > 0; off >>= 1) ce += __shfl_xor(ce, off, 32);
    if (lane == 0) ce_part[b * 8 + w] = ce;
}

// ---------------- deterministic final reduce ----------------
__global__ void finalize_kernel(const float* __restrict__ ce_part,
                                float* __restrict__ out, int n) {
    __shared__ float red[256];
    float s = 0.f;
    for (int i = threadIdx.x; i < n; i += 256) s += ce_part[i];
    red[threadIdx.x] = s;
    __syncthreads();
    for (int o = 128; o > 0; o >>= 1) {
        if ((int)threadIdx.x < o) red[threadIdx.x] += red[threadIdx.x + o];
        __syncthreads();
    }
    if (threadIdx.x == 0)
        out[0] = out[0] + WEIGHT * (red[0] / (float)(NB * GSZ));
}

extern "C" void kernel_launch(void* const* d_in, const int* in_sizes, int n_in,
                              void* d_out, int out_size, void* d_ws, size_t ws_size,
                              hipStream_t stream) {
    (void)in_sizes; (void)n_in; (void)out_size; (void)ws_size;
    const float* preds = (const float*)d_in[0];
    const float* y     = (const float*)d_in[1];
    const float* rfeat = (const float*)d_in[2];
    const float* pfeat = (const float*)d_in[3];
    // d_in[4], d_in[5]: int64 batch indices — groups are uniform & sorted, unused.
    float* out = (float*)d_out;
    float* ws  = (float*)d_ws;
    float* invr = ws;                      // NB*GSZ floats
    float* invp = ws + NB * GSZ;           // NB*GSZ floats
    float* cep  = ws + 2 * NB * GSZ;       // NB*8 floats

    mse_kernel<<<1, 256, 0, stream>>>(preds, y, out, NB);
    rownorm_kernel<<<(2 * NB * GSZ) / 8, 256, 0, stream>>>(rfeat, pfeat, invr, invp);
    contrastive_kernel<<<NB, 256, 0, stream>>>(rfeat, pfeat, invr, invp, cep);
    finalize_kernel<<<1, 256, 0, stream>>>(cep, out, NB * 8);
}
